// LSTM_20143396618693
// MI455X (gfx1250) — matrix operand order repaired and checked by reading, not compiled, not run on hardware
//
#include <hip/hip_runtime.h>
#include <cstdint>
#include <cstddef>

typedef __attribute__((ext_vector_type(16))) __bf16       v16bf;
typedef __attribute__((ext_vector_type(8)))  float        v8f;
typedef __attribute__((ext_vector_type(4)))  unsigned int v4u;

#define BATCH  256
#define TSTEPS 512

__device__ __forceinline__ float sigf(float x) { return 1.0f / (1.0f + __expf(-x)); }

__global__ void f32_to_bf16_kernel(const float* __restrict__ in, __bf16* __restrict__ out, int n) {
    int i = blockIdx.x * blockDim.x + threadIdx.x;
    if (i < n) out[i] = (__bf16)in[i];
}

// WUT[n][k] = (k < in_dim) ? W[k][n] : U[k-in_dim][n];  K = in_dim + H, n in [0, fourH)
__global__ void build_wut_kernel(const float* __restrict__ W, const float* __restrict__ U,
                                 __bf16* __restrict__ WUT, int in_dim, int H, int fourH) {
    int K = in_dim + H;
    int total = K * fourH;
    int i = blockIdx.x * blockDim.x + threadIdx.x;
    if (i >= total) return;
    int n = i / K, k = i - n * K;
    float v = (k < in_dim) ? W[(size_t)k * fourH + n] : U[(size_t)(k - in_dim) * fourH + n];
    WUT[(size_t)n * K + k] = (__bf16)v;
}

// Fused LSTM layer. The recurrence is block-diagonal in batch: one workgroup owns one
// 16-row batch block across ALL H columns, so h_{t-1} lives entirely in the LDS A-buffer
// (A = [x_t ; h_{t-1}], 16 x K bf16) and the only per-step global traffic is the x-part
// read and the h-sequence streaming store. Cell state c stays in 8 VGPRs per wave for all T.
// Grid: 16 blocks of (2*H) threads => H/16 waves; wave j owns columns [16j,16j+16) of all 4 gates.
__global__ __launch_bounds__(512)
void lstm_layer_kernel(const __bf16* __restrict__ xseq,   // [B][T][in_dim] bf16
                       const __bf16* __restrict__ WUT,    // [4H][K] bf16, K = in_dim+H
                       const float*  __restrict__ bias,   // [4H] f32
                       __bf16* __restrict__ hseq,         // [B][T][H] bf16 (layers 0/1) or null
                       __bf16* __restrict__ hlast,        // [B][H] bf16 final h (layer 2) or null
                       int in_dim, int H) {
    __shared__ __bf16 Asm[16 * 520];                      // 16 rows x KPAD(<=520) bf16

    const int T     = TSTEPS;
    const int K     = in_dim + H;
    const int KPAD  = K + 8;                              // (K+8)*2 % 16 == 0 -> 16B-aligned rows
    const int b0    = blockIdx.x * 16;
    const int wave  = threadIdx.x >> 5;
    const int lane  = threadIdx.x & 31;
    const int j0    = wave * 16;
    const int col   = lane & 15;                          // A-row / B-col / D-col for this lane
    const int khalf = (lane < 16) ? 0 : 8;                // A lane-half K offset
    const int kbase = (lane < 16) ? 0 : 8;                // B lane-half K offset
    const int mo    = (lane < 16) ? 0 : 8;                // D row offset
    const int nthr  = blockDim.x;

    const float bI = bias[0 * H + j0 + col];
    const float bF = bias[1 * H + j0 + col];
    const float bG = bias[2 * H + j0 + col];
    const float bO = bias[3 * H + j0 + col];

    const __bf16* wI = WUT + (size_t)(0 * H + j0 + col) * K;
    const __bf16* wF = WUT + (size_t)(1 * H + j0 + col) * K;
    const __bf16* wG = WUT + (size_t)(2 * H + j0 + col) * K;
    const __bf16* wO = WUT + (size_t)(3 * H + j0 + col) * K;

    v8f cst;                                              // cell state: VGPR-resident for all T
#pragma unroll
    for (int e = 0; e < 8; e++) cst[e] = 0.f;

    // x staging geometry: in_dim/2 and H/2 are powers of two -> shift/mask, no division
    const int xsh   = __builtin_ctz(in_dim >> 1);
    const int xmask = (in_dim >> 1) - 1;
    const int xTotal = 16 << xsh;

    // zero the h-region of A once (h_{-1} = 0)
    {
        const int hsh = __builtin_ctz(H >> 1);
        const int hmask = (H >> 1) - 1;
        for (int idx = threadIdx.x; idx < (16 << hsh); idx += nthr) {
            int r  = idx >> hsh;
            int kp = (idx & hmask) << 1;
            *(unsigned*)&Asm[r * KPAD + in_dim + kp] = 0u;
        }
    }

    for (int t = 0; t < T; ++t) {
        // ---- refresh x-part of A (16 x in_dim) from global; h-part already in LDS ----
        for (int idx = threadIdx.x; idx < xTotal; idx += nthr) {
            int r  = idx >> xsh;
            int kp = (idx & xmask) << 1;
            const __bf16* sp = xseq + ((size_t)(b0 + r) * T + t) * in_dim + kp;
            unsigned val = *(const unsigned*)sp;
            if (t + 1 < T) __builtin_prefetch((const void*)(sp + in_dim), 0, 1);
            *(unsigned*)&Asm[r * KPAD + kp] = val;
        }
        __syncthreads();   // x visible; previous step's h writes visible

        // ---- 4-gate GEMM over K with bf16 WMMA, f32 accumulate, bias pre-seeded ----
        v8f zI, zF, zG, zO;
#pragma unroll
        for (int e = 0; e < 8; e++) { zI[e] = bI; zF[e] = bF; zG[e] = bG; zO[e] = bO; }

        for (int k0 = 0; k0 < K; k0 += 32) {
            const __bf16* arow = &Asm[col * KPAD + k0 + khalf];
            v16bf a;
            ((v4u*)&a)[0] = *(const v4u*)(arow);
            ((v4u*)&a)[1] = *(const v4u*)(arow + 16);

            v16bf bIv, bFv, bGv, bOv;
            const __bf16* p;
            p = wI + k0 + kbase; ((v4u*)&bIv)[0] = *(const v4u*)p; ((v4u*)&bIv)[1] = *(const v4u*)(p + 16);
            p = wF + k0 + kbase; ((v4u*)&bFv)[0] = *(const v4u*)p; ((v4u*)&bFv)[1] = *(const v4u*)(p + 16);
            p = wG + k0 + kbase; ((v4u*)&bGv)[0] = *(const v4u*)p; ((v4u*)&bGv)[1] = *(const v4u*)(p + 16);
            p = wO + k0 + kbase; ((v4u*)&bOv)[0] = *(const v4u*)p; ((v4u*)&bOv)[1] = *(const v4u*)(p + 16);

            zI = __builtin_amdgcn_wmma_f32_16x16x32_bf16(false, a, false, bIv, (short)0, zI, false, false);
            zF = __builtin_amdgcn_wmma_f32_16x16x32_bf16(false, a, false, bFv, (short)0, zF, false, false);
            zG = __builtin_amdgcn_wmma_f32_16x16x32_bf16(false, a, false, bGv, (short)0, zG, false, false);
            zO = __builtin_amdgcn_wmma_f32_16x16x32_bf16(false, a, false, bOv, (short)0, zO, false, false);
        }
        __syncthreads();   // all waves done reading A before h-region is overwritten

        // ---- gates, state update; h -> LDS (next step) + global stream (next layer) ----
#pragma unroll
        for (int e = 0; e < 8; e++) {
            float iv = sigf(zI[e]);
            float fv = sigf(zF[e]);
            float gv = tanhf(zG[e]);
            float ov = sigf(zO[e]);
            float cv = fv * cst[e] + iv * gv;
            cst[e] = cv;
            float hv = ov * tanhf(cv);
            __bf16 hb = (__bf16)hv;
            int r = e + mo;                                   // row within batch block
            Asm[r * KPAD + in_dim + j0 + col] = hb;           // h_t into A for step t+1
            if (hseq) {
                hseq[((size_t)(b0 + r) * T + t) * H + j0 + col] = hb;
            } else if (t == T - 1) {
                hlast[(size_t)(b0 + r) * H + j0 + col] = hb;
            }
        }
    }
}

// out[256][64] = h[256][128] @ Wd + bd, via bf16 WMMA. 64 waves, one 16x16 tile each.
__global__ __launch_bounds__(256)
void dense_kernel(const __bf16* __restrict__ h, const __bf16* __restrict__ WdT,
                  const float* __restrict__ bd, float* __restrict__ out) {
    int wid  = blockIdx.x * 8 + (threadIdx.x >> 5);
    int lane = threadIdx.x & 31;
    int b0 = (wid >> 2) * 16, c0 = (wid & 3) * 16;
    int col   = lane & 15;
    int khalf = (lane < 16) ? 0 : 8;
    int kbase = (lane < 16) ? 0 : 8;
    int mo    = (lane < 16) ? 0 : 8;

    float bias = bd[c0 + col];
    v8f acc;
#pragma unroll
    for (int e = 0; e < 8; e++) acc[e] = bias;

#pragma unroll
    for (int k0 = 0; k0 < 128; k0 += 32) {
        v16bf a, w;
        const __bf16* ap = h + (size_t)(b0 + col) * 128 + k0 + khalf;
        ((v4u*)&a)[0] = *(const v4u*)ap; ((v4u*)&a)[1] = *(const v4u*)(ap + 16);
        const __bf16* wp = WdT + (size_t)(c0 + col) * 128 + k0 + kbase;
        ((v4u*)&w)[0] = *(const v4u*)wp; ((v4u*)&w)[1] = *(const v4u*)(wp + 16);
        acc = __builtin_amdgcn_wmma_f32_16x16x32_bf16(false, a, false, w, (short)0, acc, false, false);
    }
#pragma unroll
    for (int e = 0; e < 8; e++)
        out[(size_t)(b0 + e + mo) * 64 + c0 + col] = acc[e];
}

extern "C" void kernel_launch(void* const* d_in, const int* in_sizes, int n_in,
                              void* d_out, int out_size, void* d_ws, size_t ws_size,
                              hipStream_t stream) {
    (void)in_sizes; (void)n_in; (void)out_size; (void)ws_size;
    const float* x  = (const float*)d_in[0];
    const float* W0 = (const float*)d_in[1];
    const float* U0 = (const float*)d_in[2];
    const float* b0 = (const float*)d_in[3];
    const float* W1 = (const float*)d_in[4];
    const float* U1 = (const float*)d_in[5];
    const float* b1 = (const float*)d_in[6];
    const float* W2 = (const float*)d_in[7];
    const float* U2 = (const float*)d_in[8];
    const float* b2 = (const float*)d_in[9];
    const float* Wd = (const float*)d_in[10];
    const float* bd = (const float*)d_in[11];
    float* out = (float*)d_out;

    char* ws = (char*)d_ws;
    size_t off = 0;
    auto take = [&](size_t bytes) -> char* {
        char* p = ws + off;
        off += (bytes + 255) & ~(size_t)255;
        return p;
    };
    __bf16* x_bf  = (__bf16*)take((size_t)BATCH * TSTEPS * 64 * 2);
    __bf16* hseq0 = (__bf16*)take((size_t)BATCH * TSTEPS * 256 * 2);
    __bf16* hseq1 = (__bf16*)take((size_t)BATCH * TSTEPS * 256 * 2);
    __bf16* hlast = (__bf16*)take((size_t)BATCH * 128 * 2);
    __bf16* WUT0  = (__bf16*)take((size_t)1024 * 320 * 2);
    __bf16* WUT1  = (__bf16*)take((size_t)1024 * 512 * 2);
    __bf16* WUT2  = (__bf16*)take((size_t)512 * 384 * 2);
    __bf16* WdT   = (__bf16*)take((size_t)64 * 128 * 2);

    {
        int n = BATCH * TSTEPS * 64;
        f32_to_bf16_kernel<<<(n + 255) / 256, 256, 0, stream>>>(x, x_bf, n);
    }
    build_wut_kernel<<<(320 * 1024 + 255) / 256, 256, 0, stream>>>(W0, U0, WUT0, 64, 256, 1024);
    build_wut_kernel<<<(512 * 1024 + 255) / 256, 256, 0, stream>>>(W1, U1, WUT1, 256, 256, 1024);
    build_wut_kernel<<<(384 * 512 + 255) / 256, 256, 0, stream>>>(W2, U2, WUT2, 256, 128, 512);
    build_wut_kernel<<<(128 * 64 + 255) / 256, 256, 0, stream>>>(Wd, Wd, WdT, 128, 0, 64);

    // 16 batch blocks; one workgroup per block owns the full recurrence (no grid sync).
    lstm_layer_kernel<<<16, 512, 0, stream>>>(x_bf,  WUT0, b0, hseq0, nullptr, 64, 256);
    lstm_layer_kernel<<<16, 512, 0, stream>>>(hseq0, WUT1, b1, hseq1, nullptr, 256, 256);
    lstm_layer_kernel<<<16, 256, 0, stream>>>(hseq1, WUT2, b2, nullptr, hlast, 256, 128);

    dense_kernel<<<8, 256, 0, stream>>>(hlast, WdT, bd, out);
}
